// Airnet_65386582114401
// MI455X (gfx1250) — matrix operand order repaired and checked by reading, not compiled, not run on hardware
//
#include <hip/hip_runtime.h>
#include <hip/hip_bf16.h>

// ---------------------------------------------------------------------------
// Gated-RNN (MGU) scan over B=256 steps; per step a (512 x 2048 x 1280) GEMM
// + gates, then a final (256 x 128 x 1024) projection. bf16 WMMA, f32 accum.
// ---------------------------------------------------------------------------

typedef __attribute__((ext_vector_type(16))) __bf16 bf16x16;
typedef __attribute__((ext_vector_type(8)))  __bf16 bf16x8;
typedef __attribute__((ext_vector_type(4)))  __bf16 bf16x4;
typedef __attribute__((ext_vector_type(8)))  float  f32x8;

#define WMMA_BF16(A, B, C) \
  __builtin_amdgcn_wmma_f32_16x16x32_bf16(false, (A), false, (B), (short)0, (C), false, false)

static constexpr int BN = 256;   // scan steps (batch)
static constexpr int T  = 512;   // rows of h ("inner batch")
static constexpr int I  = 256;   // input features
static constexpr int H  = 1024;  // hidden
static constexpr int O  = 128;   // output
static constexpr int KC = I + H; // 1280 fused K

// ---- workspace layout (bytes) ----
static constexpr size_t OFF_XBF   = 0;                              // B*T*I bf16
static constexpr size_t SZ_XBF    = (size_t)BN * T * I * 2;
static constexpr size_t OFF_WCF   = OFF_XBF + SZ_XBF;               // (H x KC) bf16
static constexpr size_t SZ_WC     = (size_t)H * KC * 2;
static constexpr size_t OFF_WCH   = OFF_WCF + SZ_WC;                // (H x KC) bf16
static constexpr size_t OFF_WOUT  = OFF_WCH + SZ_WC;                // (O x H) bf16
static constexpr size_t SZ_WOUT   = (size_t)O * H * 2;
static constexpr size_t OFF_BCF   = OFF_WOUT + SZ_WOUT;             // H f32
static constexpr size_t OFF_BXH   = OFF_BCF + 4096;                 // H f32
static constexpr size_t OFF_BHH   = OFF_BXH + 4096;                 // H f32
static constexpr size_t OFF_HF    = OFF_BHH + 4096;                 // 2 * (T*H) f32 ping-pong
static constexpr size_t SZ_HF     = (size_t)T * H * 4;
static constexpr size_t OFF_HB    = OFF_HF + 2 * SZ_HF;             // 2 * (T*H) bf16 ping-pong
static constexpr size_t SZ_HB     = (size_t)T * H * 2;
static constexpr size_t OFF_LASTS = OFF_HB + 2 * SZ_HB;             // (BN x H) bf16

static __device__ __forceinline__ f32x8 splat8(float x) {
  f32x8 v;
#pragma unroll
  for (int i = 0; i < 8; ++i) v[i] = x;
  return v;
}

// ---------------------------------------------------------------------------
// Prep 1: inputs f32 -> bf16 (whole tensor, one shot)
// ---------------------------------------------------------------------------
__global__ void prep_x_kernel(const float* __restrict__ in, __bf16* __restrict__ out) {
  const size_t gid  = (size_t)blockIdx.x * blockDim.x + threadIdx.x;
  const size_t base = gid * 4;
  float4 v = ((const float4*)in)[gid];
  bf16x4 o;
  o[0] = (__bf16)v.x; o[1] = (__bf16)v.y; o[2] = (__bf16)v.z; o[3] = (__bf16)v.w;
  *(bf16x4*)(out + base) = o;
}

// ---------------------------------------------------------------------------
// Prep 2: fused weights  WcF[n,k] = k<I ? WihF[n,k] : WhhF[n,k-I]  (bf16)
//                        WcH[n,k] = k<I ? WihH[n,k] : WhhH[n,k-I]
// ---------------------------------------------------------------------------
__global__ void prep_w_kernel(const float* __restrict__ Wih,   // (2H x I)
                              const float* __restrict__ Whh,   // (2H x H)
                              __bf16* __restrict__ WcF,
                              __bf16* __restrict__ WcH) {
  const int idx = blockIdx.x * blockDim.x + threadIdx.x;       // over 2H*KC
  const int n2  = idx / KC;                                    // 0..2047
  const int k   = idx - n2 * KC;
  float v = (k < I) ? Wih[(size_t)n2 * I + k] : Whh[(size_t)n2 * H + (k - I)];
  if (n2 < H) WcF[(size_t)n2 * KC + k]       = (__bf16)v;
  else        WcH[(size_t)(n2 - H) * KC + k] = (__bf16)v;
}

// ---------------------------------------------------------------------------
// Prep 3: Wout->bf16, fused biases, zero h0 (f32 + bf16)
// ---------------------------------------------------------------------------
__global__ void prep_misc_kernel(const float* __restrict__ Wout,
                                 const float* __restrict__ Bih,
                                 const float* __restrict__ Bhh,
                                 __bf16* __restrict__ WoutBf,
                                 float* __restrict__ BcF, float* __restrict__ BxH,
                                 float* __restrict__ BhH,
                                 float* __restrict__ h0f, __bf16* __restrict__ h0b) {
  const int i = blockIdx.x * blockDim.x + threadIdx.x;         // 0 .. T*H-1
  if (i < O * H)   WoutBf[i] = (__bf16)Wout[i];
  if (i < H) {
    BcF[i] = Bih[i] + Bhh[i];
    BxH[i] = Bih[H + i];
    BhH[i] = Bhh[H + i];
  }
  h0f[i] = 0.0f;
  h0b[i] = (__bf16)0.0f;
}

// ---------------------------------------------------------------------------
// Step kernel: one recurrence step.
//   sF = [x|h] @ WcF.T + (BihF+BhhF)        (K = 1280)
//   xh =  x    @ WihH.T + BihH              (K = 256,  phase 1)
//   hh =  h    @ WhhH.T + BhhH              (K = 1024, phase 2)
//   f = sigmoid(sF); g = tanh(xh + f*hh); h' = (1-f)*h + f*g
// Grid: (16 n-tiles of 64, 8 m-tiles of 64), 128 thr = 4 waves (2x2),
// each wave computes a 32x32 tile = 2x2 WMMA tiles x 3 accumulators.
// ---------------------------------------------------------------------------
__global__ __launch_bounds__(128)
void step_kernel(const __bf16* __restrict__ xs,      // (T x I) this step, bf16
                 const __bf16* __restrict__ hb_in,   // (T x H) bf16
                 const float*  __restrict__ hf_in,   // (T x H) f32
                 __bf16* __restrict__ hb_out,
                 float*  __restrict__ hf_out,
                 const __bf16* __restrict__ WcF,     // (H x KC)
                 const __bf16* __restrict__ WcH,     // (H x KC)
                 const float* __restrict__ BcF,
                 const float* __restrict__ BxH,
                 const float* __restrict__ BhH,
                 __bf16* __restrict__ lasts_s) {     // (H) row for this step
  const int lane = threadIdx.x & 31;
  const int wave = threadIdx.x >> 5;
  const int hs   = lane >> 4;        // K-half select within fragment
  const int ln   = lane & 15;
  const int mb   = blockIdx.y * 64 + (wave >> 1) * 32;
  const int nb   = blockIdx.x * 64 + (wave & 1) * 32;

  f32x8 accF[2][2], accX[2][2], accH[2][2];
#pragma unroll
  for (int mi = 0; mi < 2; ++mi)
#pragma unroll
    for (int ni = 0; ni < 2; ++ni) {
      const int col = nb + ni * 16 + ln;
      accF[mi][ni] = splat8(BcF[col]);
      accX[mi][ni] = splat8(BxH[col]);
      accH[mi][ni] = splat8(BhH[col]);
    }

  // ---- Phase 1: k in [0, I) ; A from x ; WMMAs feed {F, XH} ----
  for (int kb = 0; kb < I; kb += 32) {
    bf16x16 A[2], BF[2], BH[2];
#pragma unroll
    for (int mi = 0; mi < 2; ++mi) {
      const int row = mb + mi * 16 + ln;
      const __bf16* p = xs + (size_t)row * I + kb + hs * 16;
      *(bf16x8*)&A[mi]       = *(const bf16x8*)p;
      *((bf16x8*)&A[mi] + 1) = *(const bf16x8*)(p + 8);
    }
#pragma unroll
    for (int ni = 0; ni < 2; ++ni) {
      const int n = nb + ni * 16 + ln;
      BF[ni] = *(const bf16x16*)(WcF + (size_t)n * KC + kb + hs * 16);
      BH[ni] = *(const bf16x16*)(WcH + (size_t)n * KC + kb + hs * 16);
    }
#pragma unroll
    for (int mi = 0; mi < 2; ++mi)
#pragma unroll
      for (int ni = 0; ni < 2; ++ni) {
        accF[mi][ni] = WMMA_BF16(A[mi], BF[ni], accF[mi][ni]);
        accX[mi][ni] = WMMA_BF16(A[mi], BH[ni], accX[mi][ni]);
      }
  }

  // ---- Phase 2: k in [I, KC) ; A from h ; WMMAs feed {F, HH} ----
  for (int kb = I; kb < KC; kb += 32) {
    const int kh = kb - I;
    bf16x16 A[2], BF[2], BH[2];
#pragma unroll
    for (int mi = 0; mi < 2; ++mi) {
      const int row = mb + mi * 16 + ln;
      const __bf16* p = hb_in + (size_t)row * H + kh + hs * 16;
      *(bf16x8*)&A[mi]       = *(const bf16x8*)p;
      *((bf16x8*)&A[mi] + 1) = *(const bf16x8*)(p + 8);
    }
#pragma unroll
    for (int ni = 0; ni < 2; ++ni) {
      const int n = nb + ni * 16 + ln;
      BF[ni] = *(const bf16x16*)(WcF + (size_t)n * KC + kb + hs * 16);
      BH[ni] = *(const bf16x16*)(WcH + (size_t)n * KC + kb + hs * 16);
    }
#pragma unroll
    for (int mi = 0; mi < 2; ++mi)
#pragma unroll
      for (int ni = 0; ni < 2; ++ni) {
        accF[mi][ni] = WMMA_BF16(A[mi], BF[ni], accF[mi][ni]);
        accH[mi][ni] = WMMA_BF16(A[mi], BH[ni], accH[mi][ni]);
      }
  }

  // ---- Epilogue: gates + state update ----
#pragma unroll
  for (int mi = 0; mi < 2; ++mi)
#pragma unroll
    for (int ni = 0; ni < 2; ++ni) {
      const int col = nb + ni * 16 + ln;
#pragma unroll
      for (int r = 0; r < 8; ++r) {
        const int row = mb + mi * 16 + hs * 8 + r;
        const float sF = accF[mi][ni][r];
        const float f  = 1.0f / (1.0f + __expf(-sF));
        const float z  = accX[mi][ni][r] + f * accH[mi][ni][r];
        const float e  = __expf(-2.0f * z);
        const float g  = (1.0f - e) / (1.0f + e);          // tanh(z)
        const float ho = hf_in[(size_t)row * H + col];
        const float hn = (1.0f - f) * ho + f * g;
        hf_out[(size_t)row * H + col] = hn;
        hb_out[(size_t)row * H + col] = (__bf16)hn;
        if (row == T - 1) lasts_s[col] = (__bf16)hn;       // emit h_new[-1]
      }
    }
}

// ---------------------------------------------------------------------------
// Output projection: out = lasts(BN x H) @ Wout.T(H x O) + Bout, f32 out.
// 128 tiles of 16x16; 32 blocks x 4 waves, one tile per wave.
// ---------------------------------------------------------------------------
__global__ __launch_bounds__(128)
void out_kernel(const __bf16* __restrict__ lasts,   // (BN x H)
                const __bf16* __restrict__ WoutBf,  // (O x H)
                const float* __restrict__ Bout,
                float* __restrict__ out) {          // (BN x O)
  const int lane = threadIdx.x & 31;
  const int wave = threadIdx.x >> 5;
  const int hs   = lane >> 4;
  const int ln   = lane & 15;
  const int wid  = blockIdx.x * 4 + wave;           // 0..127
  const int tm   = wid >> 3;                        // 0..15
  const int tn   = wid & 7;                         // 0..7

  f32x8 acc = splat8(0.0f);
  for (int kb = 0; kb < H; kb += 32) {
    bf16x16 A, Bm;
    const int row = tm * 16 + ln;
    const __bf16* p = lasts + (size_t)row * H + kb + hs * 16;
    *(bf16x8*)&A       = *(const bf16x8*)p;
    *((bf16x8*)&A + 1) = *(const bf16x8*)(p + 8);
    const int n = tn * 16 + ln;
    Bm = *(const bf16x16*)(WoutBf + (size_t)n * H + kb + hs * 16);
    acc = WMMA_BF16(A, Bm, acc);
  }
  const int col = tn * 16 + ln;
#pragma unroll
  for (int r = 0; r < 8; ++r) {
    const int row = tm * 16 + hs * 8 + r;
    out[(size_t)row * O + col] = acc[r] + Bout[col];
  }
}

// ---------------------------------------------------------------------------
extern "C" void kernel_launch(void* const* d_in, const int* in_sizes, int n_in,
                              void* d_out, int out_size, void* d_ws, size_t ws_size,
                              hipStream_t stream) {
  const float* inputs = (const float*)d_in[0];   // (BN, T, I)
  const float* Wih    = (const float*)d_in[1];   // (2H, I)
  const float* Whh    = (const float*)d_in[2];   // (2H, H)
  const float* Bih    = (const float*)d_in[3];   // (2H)
  const float* Bhh    = (const float*)d_in[4];   // (2H)
  const float* Wout   = (const float*)d_in[5];   // (O, H)
  const float* Bout   = (const float*)d_in[6];   // (O)

  char* ws = (char*)d_ws;
  __bf16* xbf    = (__bf16*)(ws + OFF_XBF);
  __bf16* WcF    = (__bf16*)(ws + OFF_WCF);
  __bf16* WcH    = (__bf16*)(ws + OFF_WCH);
  __bf16* WoutBf = (__bf16*)(ws + OFF_WOUT);
  float*  BcF    = (float*)(ws + OFF_BCF);
  float*  BxH    = (float*)(ws + OFF_BXH);
  float*  BhH    = (float*)(ws + OFF_BHH);
  float*  hf[2]  = {(float*)(ws + OFF_HF), (float*)(ws + OFF_HF + SZ_HF)};
  __bf16* hb[2]  = {(__bf16*)(ws + OFF_HB), (__bf16*)(ws + OFF_HB + SZ_HB)};
  __bf16* lasts  = (__bf16*)(ws + OFF_LASTS);

  // Prep (parallel, one-shot)
  prep_x_kernel<<<(BN * T * I) / (256 * 4), 256, 0, stream>>>(inputs, xbf);
  prep_w_kernel<<<(2 * H * KC) / 256, 256, 0, stream>>>(Wih, Whh, WcF, WcH);
  prep_misc_kernel<<<(T * H) / 256, 256, 0, stream>>>(Wout, Bih, Bhh, WoutBf,
                                                      BcF, BxH, BhH, hf[0], hb[0]);

  // Sequential recurrence: 256 dependent step-GEMMs (ping-pong h buffers)
  for (int s = 0; s < BN; ++s) {
    const int pi = s & 1, po = pi ^ 1;
    step_kernel<<<dim3(H / 64, T / 64), 128, 0, stream>>>(
        xbf + (size_t)s * T * I, hb[pi], hf[pi], hb[po], hf[po],
        WcF, WcH, BcF, BxH, BhH, lasts + (size_t)s * H);
  }

  // Output projection
  out_kernel<<<32, 128, 0, stream>>>(lasts, WoutBf, Bout, (float*)d_out);
}
